// Attention_62156766708341
// MI455X (gfx1250) — matrix operand order repaired and checked by reading, not compiled, not run on hardware
//
#include <hip/hip_runtime.h>

typedef _Float16 v16h __attribute__((ext_vector_type(16)));
typedef _Float16 v8h  __attribute__((ext_vector_type(8)));
typedef _Float16 v4h  __attribute__((ext_vector_type(4)));
typedef _Float16 v2h  __attribute__((ext_vector_type(2)));
typedef float    v8f  __attribute__((ext_vector_type(8)));

#define NH     12
#define HD     64
#define SEQ    2048
#define NF     768
#define QKV    2304   // 3*NF

#define WAVES   8
#define QTILE   16
#define QT_PER_WAVE 2
#define BLOCK_Q (WAVES * QTILE * QT_PER_WAVE)   // 256 query rows per block
#define KB      64                              // keys staged per LDS block
#define PAD     8

// 1/sqrt(64) * log2(e): base-2 softmax with scale folded into Q
#define SCALE_L2E 0.180336880111120426f

static __device__ __forceinline__ v16h join8(v8h lo, v8h hi) {
    return __builtin_shufflevector(lo, hi, 0, 1, 2, 3, 4, 5, 6, 7,
                                           8, 9, 10, 11, 12, 13, 14, 15);
}

static __device__ __forceinline__ float exp2_fast(float x) {
#if __has_builtin(__builtin_amdgcn_exp2f)
    return __builtin_amdgcn_exp2f(x);
#else
    return exp2f(x);
#endif
}

static __device__ __forceinline__ v2h pk_f16(float a, float b) {
#if __has_builtin(__builtin_amdgcn_cvt_pkrtz)
    return __builtin_bit_cast(v2h, __builtin_amdgcn_cvt_pkrtz(a, b));
#else
    v2h r = { (_Float16)a, (_Float16)b };
    return r;
#endif
}

// max-reduce within each 16-lane half via VALU lane permutes (no DS traffic)
static __device__ __forceinline__ float redmax16(float x) {
#if __has_builtin(__builtin_amdgcn_permlane16)
    int v, y;
    v = __float_as_int(x);
    y = __builtin_amdgcn_permlane16(v, v, (int)0x67452301, (int)0xEFCDAB89, false, false); // xor 1
    x = fmaxf(x, __int_as_float(y));
    v = __float_as_int(x);
    y = __builtin_amdgcn_permlane16(v, v, (int)0x54761032, (int)0xDCFE98BA, false, false); // xor 2
    x = fmaxf(x, __int_as_float(y));
    v = __float_as_int(x);
    y = __builtin_amdgcn_permlane16(v, v, (int)0x32107654, (int)0xBA98FEDC, false, false); // xor 4
    x = fmaxf(x, __int_as_float(y));
    v = __float_as_int(x);
    y = __builtin_amdgcn_permlane16(v, v, (int)0xFEDCBA98, (int)0x76543210, false, false); // xor 8
    x = fmaxf(x, __int_as_float(y));
    return x;
#else
    x = fmaxf(x, __shfl_xor(x, 1, 32));
    x = fmaxf(x, __shfl_xor(x, 2, 32));
    x = fmaxf(x, __shfl_xor(x, 4, 32));
    x = fmaxf(x, __shfl_xor(x, 8, 32));
    return x;
#endif
}

// Load one Q A-fragment (16x32 f16) directly from global: lane owns row M=lane%16,
// halves = two contiguous 8-d runs at c0 and c1 (ISA 7.12.2 A layout).
static __device__ __forceinline__ v16h load_qfrag(const float* qrow, int c0, int c1) {
    float4 a = *(const float4*)&qrow[c0];
    float4 b = *(const float4*)&qrow[c0 + 4];
    float4 c = *(const float4*)&qrow[c1];
    float4 d = *(const float4*)&qrow[c1 + 4];
    v16h r = { (_Float16)(a.x * SCALE_L2E), (_Float16)(a.y * SCALE_L2E),
               (_Float16)(a.z * SCALE_L2E), (_Float16)(a.w * SCALE_L2E),
               (_Float16)(b.x * SCALE_L2E), (_Float16)(b.y * SCALE_L2E),
               (_Float16)(b.z * SCALE_L2E), (_Float16)(b.w * SCALE_L2E),
               (_Float16)(c.x * SCALE_L2E), (_Float16)(c.y * SCALE_L2E),
               (_Float16)(c.z * SCALE_L2E), (_Float16)(c.w * SCALE_L2E),
               (_Float16)(d.x * SCALE_L2E), (_Float16)(d.y * SCALE_L2E),
               (_Float16)(d.z * SCALE_L2E), (_Float16)(d.w * SCALE_L2E) };
    return r;
}

__global__ __launch_bounds__(256)
void attn_flash_wmma(const float* __restrict__ inp, float* __restrict__ out) {
    // row strides: 72 halves = 144B -> multiples of 16B
    __shared__ alignas(16) _Float16 sKt[HD][KB + PAD];   // row-major: [key][d]
    __shared__ alignas(16) _Float16 sV[KB][HD + PAD];    // transposed: [d][P-permuted key]
    __shared__ alignas(16) _Float16 sP[WAVES][QT_PER_WAVE][QTILE][KB + PAD];

    const int tid  = threadIdx.x;
    const int wave = tid >> 5;
    const int lane = tid & 31;

    const int bid = blockIdx.x;
    const int qt  = bid & 7;         // 2048/256 = 8 q-blocks
    const int bh  = bid >> 3;
    const int h   = bh % NH;
    const int b   = bh / NH;

    const int qbase = qt * BLOCK_Q + wave * (QTILE * QT_PER_WAVE);

    const float* qptr = inp + (size_t)b * SEQ * QKV + h * HD;
    const float* kptr = qptr + NF;
    const float* vptr = qptr + 2 * NF;

    // A-fragment halves are contiguous: K = b0..b0+7 and b1..b1+7 (ISA 7.12.2)
    const int b0 = (lane < 16) ? 0 : 8;
    const int b1 = (lane < 16) ? 16 : 24;
    const int am = lane & 15;                // A-fragment row M = lane % 16

    // ---- Q tiles straight into registers (no LDS) ----
    v16h aq0[QT_PER_WAVE], aq1[QT_PER_WAVE];
    #pragma unroll
    for (int t = 0; t < QT_PER_WAVE; ++t) {
        const float* qrow = qptr + (size_t)(qbase + t * QTILE + am) * QKV;
        aq0[t] = load_qfrag(qrow, b0, b1);           // d 0..31
        aq1[t] = load_qfrag(qrow, 32 + b0, 32 + b1); // d 32..63
    }

    // all-ones B fragment: rowsum(P) via WMMA (every C column = row sum)
    const v16h bones = { (_Float16)1, (_Float16)1, (_Float16)1, (_Float16)1,
                         (_Float16)1, (_Float16)1, (_Float16)1, (_Float16)1,
                         (_Float16)1, (_Float16)1, (_Float16)1, (_Float16)1,
                         (_Float16)1, (_Float16)1, (_Float16)1, (_Float16)1 };

    // ---- online-softmax state (rows r and r+8 split across lane halves) ----
    float mrow[QT_PER_WAVE][8], lrow[QT_PER_WAVE][8];
    v8f acc[QT_PER_WAVE][4];
    #pragma unroll
    for (int t = 0; t < QT_PER_WAVE; ++t) {
        #pragma unroll
        for (int r = 0; r < 8; ++r) { mrow[t][r] = -1e30f; lrow[t][r] = 0.0f; }
        #pragma unroll
        for (int c = 0; c < 4; ++c) { v8f z = {}; acc[t][c] = z; }
    }

    const int prow = (lane < 16) ? 0 : 8;
    const int pcol = lane & 15;
    // K-permuted V row for the P*V contraction: key = (kk&1)*16 + (kk>>1), kk = lane
    const int vperm = (lane & 1) * 16 + (lane >> 1);

    for (int kb = 0; kb < SEQ; kb += KB) {
        __syncthreads();
        // cooperative K/V stage: 64 keys x 64 dims; 256 threads x 4 float4 each
        #pragma unroll
        for (int i = 0; i < 4; ++i) {
            int idx = i * 256 + tid;         // float4 index
            int key = idx >> 4;
            int c = (idx & 15) * 4;
            size_t off = (size_t)(kb + key) * QKV + c;
            float4 fk = *(const float4*)&kptr[off];
            float4 fv = *(const float4*)&vptr[off];
            // K: row-major packed b64 store [key][d]
            v4h hk = { (_Float16)fk.x, (_Float16)fk.y, (_Float16)fk.z, (_Float16)fk.w };
            *(v4h*)&sKt[key][c] = hk;
            // V: transpose into [d][pos], pos = P-permuted key (4 b16 stores)
            int vp = (key & 32) | ((key & 15) << 1) | ((key >> 4) & 1);
            sV[c + 0][vp] = (_Float16)fv.x;
            sV[c + 1][vp] = (_Float16)fv.y;
            sV[c + 2][vp] = (_Float16)fv.z;
            sV[c + 3][vp] = (_Float16)fv.w;
        }
        __syncthreads();

        // ---- S phase: full 16x64 score tile per query tile (16 WMMAs) ----
        v8f sc[QT_PER_WAVE][4];
        #pragma unroll
        for (int kt = 0; kt < 4; ++kt) {           // 16-key sub-tiles
            const v16h bk0 = join8(*(const v8h*)&sKt[kt * 16 + pcol][b0],
                                   *(const v8h*)&sKt[kt * 16 + pcol][b1]);
            const v16h bk1 = join8(*(const v8h*)&sKt[kt * 16 + pcol][32 + b0],
                                   *(const v8h*)&sKt[kt * 16 + pcol][32 + b1]);
            #pragma unroll
            for (int t = 0; t < QT_PER_WAVE; ++t) {
                v8f z = {};
                z = __builtin_amdgcn_wmma_f32_16x16x32_f16(false, aq0[t], false, bk0, (short)0, z, false, false);
                z = __builtin_amdgcn_wmma_f32_16x16x32_f16(false, aq1[t], false, bk1, (short)0, z, false, false);
                sc[t][kt] = z;
            }
        }

        // ---- one softmax pass over all 64 key columns ----
        float corr[QT_PER_WAVE][8];
        #pragma unroll
        for (int t = 0; t < QT_PER_WAVE; ++t) {
            #pragma unroll
            for (int r = 0; r < 8; ++r) {
                float x0 = sc[t][0][r];
                float x1 = sc[t][1][r];
                float x2 = sc[t][2][r];
                float x3 = sc[t][3][r];
                float mx = redmax16(fmaxf(fmaxf(x0, x1), fmaxf(x2, x3)));
                float mnew = fmaxf(mrow[t][r], mx);
                float p0 = exp2_fast(x0 - mnew);
                float p1 = exp2_fast(x1 - mnew);
                float p2 = exp2_fast(x2 - mnew);
                float p3 = exp2_fast(x3 - mnew);
                corr[t][r] = exp2_fast(mrow[t][r] - mnew);
                mrow[t][r] = mnew;
                // K-permuted pair packing: keys j and j+16 adjacent -> one b32 store
                *(v2h*)&sP[wave][t][prow + r][2 * pcol]      = pk_f16(p0, p1);
                *(v2h*)&sP[wave][t][prow + r][32 + 2 * pcol] = pk_f16(p2, p3);
            }
        }

        // same-wave DS ops are in order: safe to read sP back now (b128 loads)
        v16h ap0[QT_PER_WAVE], ap1[QT_PER_WAVE];
        #pragma unroll
        for (int t = 0; t < QT_PER_WAVE; ++t) {
            ap0[t] = join8(*(const v8h*)&sP[wave][t][am][b0],
                           *(const v8h*)&sP[wave][t][am][b1]);
            ap1[t] = join8(*(const v8h*)&sP[wave][t][am][32 + b0],
                           *(const v8h*)&sP[wave][t][am][32 + b1]);
        }

        // row sums via matrix engine + single state update per 64 keys
        #pragma unroll
        for (int t = 0; t < QT_PER_WAVE; ++t) {
            v8f zs = {};
            v8f ssum = __builtin_amdgcn_wmma_f32_16x16x32_f16(
                false, ap0[t], false, bones, (short)0, zs, false, false);
            ssum = __builtin_amdgcn_wmma_f32_16x16x32_f16(
                false, ap1[t], false, bones, (short)0, ssum, false, false);
            #pragma unroll
            for (int r = 0; r < 8; ++r) {
                lrow[t][r] = lrow[t][r] * corr[t][r] + ssum[r];
                acc[t][0][r] *= corr[t][r];
                acc[t][1][r] *= corr[t][r];
                acc[t][2][r] *= corr[t][r];
                acc[t][3][r] *= corr[t][r];
            }
        }

        // ---- O += P(16x64) * V(64x64): V fragments shared by both query tiles ----
        #pragma unroll
        for (int c = 0; c < 4; ++c) {
            const v16h bv0 = join8(*(const v8h*)&sV[c * 16 + pcol][b0],
                                   *(const v8h*)&sV[c * 16 + pcol][b1]);
            const v16h bv1 = join8(*(const v8h*)&sV[c * 16 + pcol][32 + b0],
                                   *(const v8h*)&sV[c * 16 + pcol][32 + b1]);
            #pragma unroll
            for (int t = 0; t < QT_PER_WAVE; ++t) {
                acc[t][c] = __builtin_amdgcn_wmma_f32_16x16x32_f16(
                    false, ap0[t], false, bv0, (short)0, acc[t][c], false, false);
                acc[t][c] = __builtin_amdgcn_wmma_f32_16x16x32_f16(
                    false, ap1[t], false, bv1, (short)0, acc[t][c], false, false);
            }
        }
    }

    // ---- epilogue: normalize and store fp32 ----
    #pragma unroll
    for (int t = 0; t < QT_PER_WAVE; ++t) {
        #pragma unroll
        for (int r = 0; r < 8; ++r) {
            float inv = 1.0f / lrow[t][r];
            int row = qbase + t * QTILE + prow + r;
            float* o = out + ((size_t)b * SEQ + row) * NF + h * HD;
            o[0 * 16 + pcol] = acc[t][0][r] * inv;
            o[1 * 16 + pcol] = acc[t][1][r] * inv;
            o[2 * 16 + pcol] = acc[t][2][r] * inv;
            o[3 * 16 + pcol] = acc[t][3][r] * inv;
        }
    }
}

extern "C" void kernel_launch(void* const* d_in, const int* in_sizes, int n_in,
                              void* d_out, int out_size, void* d_ws, size_t ws_size,
                              hipStream_t stream) {
    const float* inp = (const float*)d_in[0];
    float* out = (float*)d_out;
    const int blocks = 4 * NH * (SEQ / BLOCK_Q);   // 384
    attn_flash_wmma<<<blocks, 256, 0, stream>>>(inp, out);
}
